// Encoder_78984448573588
// MI455X (gfx1250) — matrix operand order repaired and checked by reading, not compiled, not run on hardware
//
#include <hip/hip_runtime.h>
#include <hip/hip_bf16.h>

// ---------------------------------------------------------------------------
// Bidirectional LSTM encoder for MI455X (gfx1250), wave32 + WMMA bf16.
// B=64, T=512, D_IN=512, H=1024, D_DEC=1024.
//  - per-block W_hh gate-slice staged into 128KB LDS via TDM (tensor_load_to_lds)
//  - x@W_ih^T GEMM software-pipelined under the inter-WGP barrier
//  - barrier: release fetch_add + acquire plain-load spin (no RMW polling)
// ---------------------------------------------------------------------------

typedef __bf16 bf16;
typedef __attribute__((ext_vector_type(16))) __bf16 bf16x16;
typedef __attribute__((ext_vector_type(8)))  __bf16 bf16x8;
typedef __attribute__((ext_vector_type(8)))  float  f32x8;
typedef __attribute__((ext_vector_type(4)))  unsigned int u32x4;
typedef __attribute__((ext_vector_type(8)))  int i32x8;
typedef __attribute__((ext_vector_type(4)))  int i32x4;

#define T_LEN  512
#define B_SZ   64
#define DIN    512
#define H_SZ   1024
#define DDEC   1024
#define NWG    64                 // workgroups per direction
#define LDS_W_BYTES (4 * 16 * H_SZ * 2)   // 128 KB: [4 gates][16 rows][1024 k] bf16

#if defined(__HIP_DEVICE_COMPILE__) && __has_builtin(__builtin_amdgcn_tensor_load_to_lds) && \
    __has_builtin(__builtin_amdgcn_s_wait_tensorcnt)
#define USE_TDM 1
#else
#define USE_TDM 0
#endif

// ----- fragment helpers -----------------------------------------------------
// A-fragment (16x32 bf16, MxK): lane l -> row M=(l&15); data = K chunks
//   [lh*8, lh*8+8) and [lh*8+16, lh*8+24) of the k-tile (two b128 loads).
// B-fragment (32x16 bf16, KxN): lane l -> col N=(l&15); data = K chunks
//   [lh*8, lh*8+8) and [lh*8+16, lh*8+24) (two b128 loads). Weights are row-major [n][k].
static __device__ __forceinline__ bf16x16 load_frag2(const bf16* p0, const bf16* p1) {
    bf16x8 lo = *(const bf16x8*)p0;
    bf16x8 hi = *(const bf16x8*)p1;
    return __builtin_shufflevector(lo, hi, 0,1,2,3,4,5,6,7,8,9,10,11,12,13,14,15);
}

static __device__ __forceinline__ f32x8 wmma_bf16(bf16x16 a, bf16x16 b, f32x8 c) {
    // (neg_a, A, neg_b, B, c_mod, C, reuse_a, reuse_b)
    return __builtin_amdgcn_wmma_f32_16x16x32_bf16(false, a, false, b, (short)0, c,
                                                   false, false);
}

static __device__ __forceinline__ float sigmoidf_(float x) {
    return 1.0f / (1.0f + __expf(-x));
}

// ----- prep: f32 -> bf16 ----------------------------------------------------
__global__ void f32_to_bf16_kernel(const float* __restrict__ src,
                                   bf16* __restrict__ dst, int n) {
    int i = blockIdx.x * blockDim.x + threadIdx.x;
    int stride = gridDim.x * blockDim.x;
    for (; i < n; i += stride) dst[i] = (bf16)src[i];
}

// ----- prep: fold biases, reset barrier counters ----------------------------
__global__ void bias_prep_kernel(const float* __restrict__ bihF, const float* __restrict__ bhhF,
                                 const float* __restrict__ bihB, const float* __restrict__ bhhB,
                                 float* __restrict__ bsum, unsigned int* __restrict__ bar) {
    int i = blockIdx.x * blockDim.x + threadIdx.x;
    if (i < 4 * H_SZ) {
        bsum[i] = bihF[i] + bhhF[i];
    } else if (i < 8 * H_SZ) {
        int j = i - 4 * H_SZ;
        bsum[i] = bihB[j] + bhhB[j];
    }
    if (i < 2) bar[i] = 0u;
}

// ----- persistent recurrent kernel ------------------------------------------
// grid: 128 blocks = {dir} x {64 column-slices}, 128 threads = 4 waves.
// Wave w owns batch rows [16w,16w+16); block owns hidden units [16wg,16wg+16).
__global__ __launch_bounds__(128) void lstm_bidir_kernel(
    const bf16* __restrict__ xbf,                          // (B, T, DIN) bf16
    const bf16* __restrict__ WihF, const bf16* __restrict__ WhhF,
    const bf16* __restrict__ WihB, const bf16* __restrict__ WhhB,
    const float* __restrict__ bsum,                        // [2][4H]
    bf16* __restrict__ hbuf,                               // [2 dir][2 phase][B][H]
    float* __restrict__ hs_out,                            // (B, T, 2H) f32 (d_out)
    bf16* __restrict__ hend, bf16* __restrict__ cend,      // (B, 2H) bf16
    unsigned int* __restrict__ bar)                        // [2]
{
    extern __shared__ __align__(16) char smem[];
    bf16* wlds = (bf16*)smem;                              // [4][16][H_SZ]

    const int blk  = blockIdx.x;
    const int dir  = blk >> 6;
    const int wg   = blk & (NWG - 1);
    const int j0   = wg * 16;
    const int wave = threadIdx.x >> 5;       // M-tile (batch/16)
    const int lane = threadIdx.x & 31;
    const int ln   = lane & 15;
    const int lh   = lane >> 4;

    const bf16* Wih = dir ? WihB : WihF;
    const bf16* Whh = dir ? WhhB : WhhF;
    const float* bias = bsum + dir * 4 * H_SZ;
    bf16* hdir = hbuf + (size_t)dir * (2 * B_SZ * H_SZ);
    unsigned int* cnt = bar + dir;

    // ---- stage this block's W_hh gate-slice into LDS (persistent) ----------
    // Rows g*H + j0 .. +16 are contiguous in global memory per gate: 4 chunks
    // of 16*1024 bf16 (32KB) each.
#if USE_TDM
    if (wave == 0) {
        #pragma unroll
        for (int g = 0; g < 4; ++g) {
            const unsigned long long ga =
                (unsigned long long)(Whh + (size_t)(g * H_SZ + j0) * H_SZ);
            const unsigned lds_off =
                (unsigned)(unsigned long long)(wlds + (size_t)g * (16 * H_SZ));
            u32x4 g0;
            g0[0] = 1u;                                       // count=1, load, user
            g0[1] = lds_off;                                  // lds_addr [63:32]
            g0[2] = (unsigned)(ga & 0xffffffffu);             // global_addr [95:64]
            g0[3] = (unsigned)((ga >> 32) & 0x1ffffffu)       // global_addr [120:96]
                    | (2u << 30);                             // type=2 (image)
            i32x8 g1 = {};
            g1[0] = (int)(1u << 16);                          // data_size=1 (2 bytes)
            g1[1] = (int)(16384u << 16);                      // tensor_dim0[15:0]=16384
            g1[2] = (int)(1u << 16);                          // dim0 hi=0 | tensor_dim1=1
            g1[3] = (int)(16384u << 16);                      // dim1 hi=0 | tile_dim0=16384
            g1[4] = 1;                                        // tile_dim1=1, tile_dim2=0
            g1[5] = 16384;                                    // tensor_dim0_stride
            g1[6] = 0;
            g1[7] = 0;
            i32x4 gz = {};
#if __clang_major__ >= 23
            i32x8 gz8 = {};
            __builtin_amdgcn_tensor_load_to_lds(g0, g1, gz, gz, gz8, 0);
#else
            __builtin_amdgcn_tensor_load_to_lds(g0, g1, gz, gz, 0);
#endif
        }
        __builtin_amdgcn_s_wait_tensorcnt(0);
    }
#else
    for (int g = 0; g < 4; ++g) {
        const bf16x8* src = (const bf16x8*)(Whh + (size_t)(g * H_SZ + j0) * H_SZ);
        bf16x8* dst = (bf16x8*)(wlds + g * (16 * H_SZ));
        #pragma unroll 4
        for (int i = threadIdx.x; i < (16 * H_SZ) / 8; i += 128) dst[i] = src[i];
    }
#endif
    __syncthreads();

    const int arow = wave * 16 + ln;         // batch row this lane loads for A

    const float bi  = bias[0 * H_SZ + j0 + ln];
    const float bf_ = bias[1 * H_SZ + j0 + ln];
    const float bg  = bias[2 * H_SZ + j0 + ln];
    const float bo  = bias[3 * H_SZ + j0 + ln];

    // global B-fragment row bases for W_ih (streamed; L2-resident, 8MB total)
    const bf16* wx0 = Wih + (size_t)(0 * H_SZ + j0 + ln) * DIN;
    const bf16* wx1 = Wih + (size_t)(1 * H_SZ + j0 + ln) * DIN;
    const bf16* wx2 = Wih + (size_t)(2 * H_SZ + j0 + ln) * DIN;
    const bf16* wx3 = Wih + (size_t)(3 * H_SZ + j0 + ln) * DIN;
    // LDS B-fragment row bases for W_hh
    const bf16* wl0 = wlds + (size_t)(0 * 16 + ln) * H_SZ;
    const bf16* wl1 = wlds + (size_t)(1 * 16 + ln) * H_SZ;
    const bf16* wl2 = wlds + (size_t)(2 * 16 + ln) * H_SZ;
    const bf16* wl3 = wlds + (size_t)(3 * 16 + ln) * H_SZ;

    const f32x8 fzero = {};
    f32x8 c = {};            // persistent cell state tile
    f32x8 hreg = {};         // last hidden state tile

    // ---- pre-compute x-part for t = 0 (pipelined thereafter) ---------------
    f32x8 nxi = {}, nxf = {}, nxg = {}, nxo = {};
    {
        const int xt0 = dir ? (T_LEN - 1) : 0;
        const bf16* xrow = xbf + (size_t)arow * (T_LEN * DIN) + (size_t)xt0 * DIN;
        #pragma unroll 4
        for (int kt = 0; kt < DIN / 32; ++kt) {
            const int ka = kt * 32 + lh * 8;
            const int kb = kt * 32 + lh * 16;
            bf16x16 a = load_frag2(xrow + ka, xrow + ka + 16);
            nxi = wmma_bf16(a, load_frag2(wx0 + ka, wx0 + ka + 16), nxi);
            nxf = wmma_bf16(a, load_frag2(wx1 + ka, wx1 + ka + 16), nxf);
            nxg = wmma_bf16(a, load_frag2(wx2 + ka, wx2 + ka + 16), nxg);
            nxo = wmma_bf16(a, load_frag2(wx3 + ka, wx3 + ka + 16), nxo);
        }
    }

    for (int t = 0; t < T_LEN; ++t) {
        f32x8 ai = nxi, af = nxf, ag = nxg, ao = nxo;

        // ---- h_{t-1} @ W_hh^T (K=1024), B-fragments from LDS; h_0==0 -> skip
        if (t > 0) {
            const bf16* hrow = hdir + (size_t)(t & 1) * (B_SZ * H_SZ)
                                    + (size_t)arow * H_SZ;
            #pragma unroll 4
            for (int kt = 0; kt < H_SZ / 32; ++kt) {
                const int ka = kt * 32 + lh * 8;
                const int kb = kt * 32 + lh * 16;
                bf16x16 a = load_frag2(hrow + ka, hrow + ka + 16);
                ai = wmma_bf16(a, load_frag2(wl0 + ka, wl0 + ka + 16), ai);
                af = wmma_bf16(a, load_frag2(wl1 + ka, wl1 + ka + 16), af);
                ag = wmma_bf16(a, load_frag2(wl2 + ka, wl2 + ka + 16), ag);
                ao = wmma_bf16(a, load_frag2(wl3 + ka, wl3 + ka + 16), ao);
            }
        }

        // ---- gate nonlinearities + state update (per-lane register tiles) --
        #pragma unroll
        for (int r = 0; r < 8; ++r) {
            float gi = sigmoidf_(ai[r] + bi);
            float gf = sigmoidf_(af[r] + bf_);
            float gg = tanhf(ag[r] + bg);
            float go = sigmoidf_(ao[r] + bo);
            float cn = gf * c[r] + gi * gg;
            c[r] = cn;
            hreg[r] = go * tanhf(cn);
        }

        // ---- write h_t: bf16 double buffer + f32 hidden_states slot t ------
        bf16* hw = hdir + (size_t)((t + 1) & 1) * (B_SZ * H_SZ);
        #pragma unroll
        for (int r = 0; r < 8; ++r) {
            const int bidx = wave * 16 + lh * 8 + r;
            hw[(size_t)bidx * H_SZ + j0 + ln] = (bf16)hreg[r];
            hs_out[(size_t)bidx * (T_LEN * 2 * H_SZ) + (size_t)t * (2 * H_SZ)
                   + (size_t)(dir * H_SZ + j0 + ln)] = hreg[r];
        }

        // ---- signal barrier, then hide next step's x-GEMM under the wait ---
        __syncthreads();
        if (threadIdx.x == 0) {
            __hip_atomic_fetch_add(cnt, 1u, __ATOMIC_RELEASE,
                                   __HIP_MEMORY_SCOPE_AGENT);
        }

        nxi = fzero; nxf = fzero; nxg = fzero; nxo = fzero;
        if (t + 1 < T_LEN) {
            const int xt2 = dir ? (T_LEN - 2 - t) : (t + 1);
            const bf16* xrow = xbf + (size_t)arow * (T_LEN * DIN) + (size_t)xt2 * DIN;
            #pragma unroll 4
            for (int kt = 0; kt < DIN / 32; ++kt) {
                const int ka = kt * 32 + lh * 8;
                const int kb = kt * 32 + lh * 16;
                bf16x16 a = load_frag2(xrow + ka, xrow + ka + 16);
                nxi = wmma_bf16(a, load_frag2(wx0 + ka, wx0 + ka + 16), nxi);
                nxf = wmma_bf16(a, load_frag2(wx1 + ka, wx1 + ka + 16), nxf);
                nxg = wmma_bf16(a, load_frag2(wx2 + ka, wx2 + ka + 16), nxg);
                nxo = wmma_bf16(a, load_frag2(wx3 + ka, wx3 + ka + 16), nxo);
            }
        }

        if (threadIdx.x == 0) {
            const unsigned tgt = (unsigned)NWG * (unsigned)(t + 1);
            while (__hip_atomic_load(cnt, __ATOMIC_ACQUIRE,
                                     __HIP_MEMORY_SCOPE_AGENT) < tgt) {
                __builtin_amdgcn_s_sleep(2);
            }
        }
        __syncthreads();
    }

    // ---- final (h_T, c_T) -> (B, 2H) bf16 staging for the FC kernels -------
    #pragma unroll
    for (int r = 0; r < 8; ++r) {
        const int bidx = wave * 16 + lh * 8 + r;
        const size_t idx = (size_t)bidx * (2 * H_SZ) + (size_t)(dir * H_SZ + j0 + ln);
        hend[idx] = (bf16)hreg[r];
        cend[idx] = (bf16)c[r];
    }
}

// ----- final FC: tanh(end @ W^T + b), 64x1024 per matrix, K = 2048 ----------
// 512 one-wave blocks: {which h/c} x {4 M-tiles} x {64 N-tiles}.
__global__ __launch_bounds__(32) void fc_kernel(
    const bf16* __restrict__ hend, const bf16* __restrict__ cend,
    const bf16* __restrict__ fhW, const bf16* __restrict__ fcW,
    const float* __restrict__ fh_b, const float* __restrict__ fc_b,
    float* __restrict__ out)
{
    const int blk = blockIdx.x;
    const int which = blk >> 8;          // 0: initial_hiddens, 1: initial_cells
    const int mt = (blk >> 6) & 3;
    const int nt = blk & 63;
    const int lane = threadIdx.x & 31;
    const int ln = lane & 15, lh = lane >> 4;

    const bf16* A = which ? cend : hend;
    const bf16* W = which ? fcW : fhW;
    const float* bias = which ? fc_b : fh_b;
    float* o = out + (size_t)B_SZ * T_LEN * 2 * H_SZ + (size_t)which * B_SZ * DDEC;

    const bf16* arow = A + (size_t)(mt * 16 + ln) * (2 * H_SZ);
    const bf16* wrow = W + (size_t)(nt * 16 + ln) * (2 * H_SZ);

    f32x8 acc = {};
    #pragma unroll 8
    for (int kt = 0; kt < (2 * H_SZ) / 32; ++kt) {
        const int ka = kt * 32 + lh * 8;
        const int kb = kt * 32 + lh * 16;
        bf16x16 a = load_frag2(arow + ka, arow + ka + 16);
        bf16x16 b = load_frag2(wrow + ka, wrow + ka + 16);
        acc = wmma_bf16(a, b, acc);
    }
    const float bv = bias[nt * 16 + ln];
    #pragma unroll
    for (int r = 0; r < 8; ++r) {
        const int bidx = mt * 16 + lh * 8 + r;
        o[(size_t)bidx * DDEC + (size_t)(nt * 16 + ln)] = tanhf(acc[r] + bv);
    }
}

// ---------------------------------------------------------------------------
extern "C" void kernel_launch(void* const* d_in, const int* in_sizes, int n_in,
                              void* d_out, int out_size, void* d_ws, size_t ws_size,
                              hipStream_t stream) {
    (void)in_sizes; (void)n_in; (void)out_size; (void)ws_size;

    const float* x      = (const float*)d_in[0];
    const float* W_ih_f = (const float*)d_in[1];
    const float* W_hh_f = (const float*)d_in[2];
    const float* b_ih_f = (const float*)d_in[3];
    const float* b_hh_f = (const float*)d_in[4];
    const float* W_ih_b = (const float*)d_in[5];
    const float* W_hh_b = (const float*)d_in[6];
    const float* b_ih_b = (const float*)d_in[7];
    const float* b_hh_b = (const float*)d_in[8];
    const float* fc_W   = (const float*)d_in[9];
    const float* fc_b   = (const float*)d_in[10];
    const float* fh_W   = (const float*)d_in[11];
    const float* fh_b   = (const float*)d_in[12];
    float* out = (float*)d_out;

    // ---- carve workspace (256B aligned) ----
    char* ws = (char*)d_ws;
    size_t off = 0;
    auto alloc = [&](size_t bytes) -> void* {
        void* p = ws + off;
        off += (bytes + 255) & ~(size_t)255;
        return p;
    };
    bf16* xbf   = (bf16*)alloc((size_t)B_SZ * T_LEN * DIN * 2);
    bf16* WihFb = (bf16*)alloc((size_t)4 * H_SZ * DIN * 2);
    bf16* WhhFb = (bf16*)alloc((size_t)4 * H_SZ * H_SZ * 2);
    bf16* WihBb = (bf16*)alloc((size_t)4 * H_SZ * DIN * 2);
    bf16* WhhBb = (bf16*)alloc((size_t)4 * H_SZ * H_SZ * 2);
    bf16* fhWb  = (bf16*)alloc((size_t)DDEC * 2 * H_SZ * 2);
    bf16* fcWb  = (bf16*)alloc((size_t)DDEC * 2 * H_SZ * 2);
    float* bsum = (float*)alloc((size_t)8 * H_SZ * 4);
    bf16* hbuf  = (bf16*)alloc((size_t)2 * 2 * B_SZ * H_SZ * 2);
    bf16* hend  = (bf16*)alloc((size_t)B_SZ * 2 * H_SZ * 2);
    bf16* cend  = (bf16*)alloc((size_t)B_SZ * 2 * H_SZ * 2);
    unsigned int* bar = (unsigned int*)alloc(256);

    // ---- prep: convert to bf16, fold biases, reset barrier ----
    auto cvt = [&](const float* s, bf16* d, int n) {
        int blocks = (n + 2047) / 2048;
        f32_to_bf16_kernel<<<blocks, 256, 0, stream>>>(s, d, n);
    };
    cvt(x,      xbf,   B_SZ * T_LEN * DIN);
    cvt(W_ih_f, WihFb, 4 * H_SZ * DIN);
    cvt(W_hh_f, WhhFb, 4 * H_SZ * H_SZ);
    cvt(W_ih_b, WihBb, 4 * H_SZ * DIN);
    cvt(W_hh_b, WhhBb, 4 * H_SZ * H_SZ);
    cvt(fh_W,   fhWb,  DDEC * 2 * H_SZ);
    cvt(fc_W,   fcWb,  DDEC * 2 * H_SZ);
    bias_prep_kernel<<<(8 * H_SZ + 255) / 256, 256, 0, stream>>>(
        b_ih_f, b_hh_f, b_ih_b, b_hh_b, bsum, bar);

    // ---- persistent bidirectional recurrence (128KB dynamic LDS/block) ----
    (void)hipFuncSetAttribute((const void*)lstm_bidir_kernel,
                              hipFuncAttributeMaxDynamicSharedMemorySize,
                              LDS_W_BYTES);
    lstm_bidir_kernel<<<2 * NWG, 128, LDS_W_BYTES, stream>>>(
        xbf, WihFb, WhhFb, WihBb, WhhBb, bsum, hbuf, out, hend, cend, bar);

    // ---- final FCs ----
    fc_kernel<<<512, 32, 0, stream>>>(hend, cend, fhWb, fcWb, fh_b, fc_b, out);
}
